// SwinSAOperator2d_86071144612138
// MI455X (gfx1250) — matrix operand order repaired and checked by reading, not compiled, not run on hardware
//
#include <hip/hip_runtime.h>
#include <math.h>

#define NHD   4
#define WSZ   4
#define SHIFT 2
#define NLAYER 4
#define BATCH 8
#define HH 256
#define WW 256
#define EMB 64
#define HID 128
#define NSUB 16                 // subdomains per axis
#define SD 16                   // subdomain spatial size
#define NWX 4                   // windows per axis
#define NWIN (BATCH*NWX*NWX)    // 128
#define NTOK 16
#define NPIX (NWIN*NTOK*SD*SD)  // 524288
#define EPSN 1e-5f

typedef __attribute__((ext_vector_type(16))) _Float16 v16h;
typedef __attribute__((ext_vector_type(8)))  _Float16 v8h;
typedef __attribute__((ext_vector_type(8)))  float    v8f;

union V16U { v16h v; v8h h[2]; };

__device__ __forceinline__ float gelu_exact(float x) {
  return 0.5f * x * (1.0f + erff(x * 0.70710678118654752f));
}

// Remap (win,tok) by adding sh_add (mod NSUB) to global subdomain coords.
__device__ __forceinline__ int remap_slab(int win, int tok, int sh_add) {
  if (sh_add == 0) return win * NTOK + tok;
  int b  = win / (NWX * NWX);
  int wy = (win / NWX) % NWX;
  int wx = win % NWX;
  int ty = tok / WSZ, tx = tok % WSZ;
  int sy = (wy * WSZ + ty + sh_add) & (NSUB - 1);
  int sx = (wx * WSZ + tx + sh_add) & (NSUB - 1);
  int win2 = (b * NWX + (sy / WSZ)) * NWX + (sx / WSZ);
  int tok2 = (sy % WSZ) * WSZ + (sx % WSZ);
  return win2 * NTOK + tok2;
}

// ---------------------------------------------------------------- embed -----
// grid coords + pointwise MLP 5 -> 128 (gelu) -> 64; write f16 in window layout
__global__ __launch_bounds__(256)
void k_embed(const float* __restrict__ v, const float* __restrict__ W1,
             const float* __restrict__ b1, const float* __restrict__ W2,
             const float* __restrict__ b2, const int* __restrict__ dsy_p,
             const int* __restrict__ dsx_p, _Float16* __restrict__ Dq) {
  __shared__ float W1l[HID * 5];
  __shared__ float W2l[EMB * HID];
  __shared__ float b1l[HID];
  __shared__ float b2l[EMB];
  int tid = threadIdx.x;
  for (int i = tid; i < HID * 5; i += 256) W1l[i] = W1[i];
  for (int i = tid; i < EMB * HID; i += 256) W2l[i] = W2[i];
  if (tid < HID) b1l[tid] = b1[tid];
  if (tid < EMB) b2l[tid] = b2[tid];
  __syncthreads();

  long gid = (long)blockIdx.x * 256 + tid;           // < NPIX
  int X  = (int)(gid & 255);
  int Yp = (int)((gid >> 8) & 255);
  int b  = (int)(gid >> 16);
  float scy = (float)dsy_p[0] / (float)HH;
  float scx = (float)dsx_p[0] / (float)WW;
  float in[5];
  in[0] = ((float)Yp + 0.5f) * scy;
  in[1] = ((float)X + 0.5f) * scx;
  long vbase = (long)b * 3 * 65536 + (long)Yp * 256 + X;
  in[2] = v[vbase];
  in[3] = v[vbase + 65536];
  in[4] = v[vbase + 2 * 65536];

  float out[EMB];
  for (int e = 0; e < EMB; ++e) out[e] = b2l[e];
  for (int o = 0; o < HID; ++o) {
    float hv = b1l[o];
    for (int c = 0; c < 5; ++c) hv += W1l[o * 5 + c] * in[c];
    hv = gelu_exact(hv);
    for (int e = 0; e < EMB; ++e) out[e] += W2l[e * HID + o] * hv;
  }

  int syi = Yp >> 4, y = Yp & 15, sxi = X >> 4, x = X & 15;
  int win = (b * NWX + (syi >> 2)) * NWX + (sxi >> 2);
  int tok = (syi & 3) * WSZ + (sxi & 3);
  _Float16* db = Dq + ((long)(win * NTOK + tok) * 256 + y * 16 + x) * EMB;
  for (int cc = 0; cc < 8; ++cc) {
    v8h pk;
    for (int e = 0; e < 8; ++e) pk[e] = (_Float16)out[cc * 8 + e];
    *(v8h*)(db + cc * 8) = pk;
  }
}

// ------------------------------------------------------------ inst-norm -----
// one block (64 threads = one channel each) per (win,tok) slab of 256 points
__global__ __launch_bounds__(64)
void k_inorm(const _Float16* __restrict__ Dq, _Float16* __restrict__ Xn,
             int sh_add) {
  int slab = blockIdx.x;          // win*16+tok
  int c = threadIdx.x;            // channel
  long base = (long)slab * 256 * EMB;
  float s = 0.f, s2 = 0.f;
  for (int sp = 0; sp < 256; ++sp) {
    float val = (float)Dq[base + (long)sp * EMB + c];
    s += val; s2 += val * val;
  }
  float mu = s * (1.0f / 256.0f);
  float var = s2 * (1.0f / 256.0f) - mu * mu;
  float inv = rsqrtf(var + EPSN);
  long dbase = (long)remap_slab(slab >> 4, slab & 15, sh_add) * 256 * EMB;
  for (int sp = 0; sp < 256; ++sp) {
    float val = (float)Dq[base + (long)sp * EMB + c];
    Xn[dbase + (long)sp * EMB + c] = (_Float16)((val - mu) * inv);
  }
}

// --------------------------------------------------- pointwise WMMA GEMM ----
// Y[p][n] = act( sum_k X[p][k] * W[n][k] + bias[n] )     (64x64 weights)
// MODE 0: store Y;  MODE 1: gelu then store Y;  MODE 2: residual D += result
// (with optional subdomain un-shift remap of the destination base).
template <int MODE>
__global__ __launch_bounds__(256)
void k_gemm(const _Float16* __restrict__ X, const float* __restrict__ Wg,
            const float* __restrict__ bias, _Float16* __restrict__ Y,
            _Float16* __restrict__ D, int sh_add) {
  __shared__ __align__(16) _Float16 Wl[EMB * EMB];   // 8 KB, f16 weights
  __shared__ float bl[EMB];
  int tid = threadIdx.x;
  for (int i = tid; i < EMB * EMB; i += 256) Wl[i] = (_Float16)Wg[i];
  if (tid < EMB) bl[tid] = bias[tid];
  __syncthreads();

  int wave = tid >> 5, lane = tid & 31;
  int half = lane >> 4, mrow = lane & 15;
  int mtile = blockIdx.x * 8 + wave;

  // A fragment: 16 pixels x 64 channels (two K blocks of 32)
  const _Float16* xb = X + ((long)mtile * 16 + mrow) * EMB;
  V16U A0, A1;
  A0.h[0] = *(const v8h*)(xb + half * 16);
  A0.h[1] = *(const v8h*)(xb + half * 16 + 8);
  A1.h[0] = *(const v8h*)(xb + 32 + half * 16);
  A1.h[1] = *(const v8h*)(xb + 32 + half * 16 + 8);

  int pbase = mtile * 16;                  // rows share (win,tok,y); row m = x
  long obase = pbase;
  if (MODE == 2) {
    int y = (pbase >> 4) & 15;
    int slab2 = remap_slab(pbase >> 12, (pbase >> 8) & 15, sh_add);
    obase = (long)slab2 * 256 + y * 16;
  }

  for (int nt = 0; nt < 4; ++nt) {
    int n = nt * 16 + (lane & 15);
    const _Float16* wr = Wl + n * EMB;     // B[k][n] = W[n][k], k contiguous
    V16U B0, B1;
    B0.h[0] = *(const v8h*)(wr + half * 16);
    B0.h[1] = *(const v8h*)(wr + half * 16 + 8);
    B1.h[0] = *(const v8h*)(wr + 32 + half * 16);
    B1.h[1] = *(const v8h*)(wr + 32 + half * 16 + 8);
    float bn = bl[n];
    v8f c;
    for (int r = 0; r < 8; ++r) c[r] = bn;
    c = __builtin_amdgcn_wmma_f32_16x16x32_f16(false, A0.v, false, B0.v,
                                               (short)0, c, false, false);
    c = __builtin_amdgcn_wmma_f32_16x16x32_f16(false, A1.v, false, B1.v,
                                               (short)0, c, false, false);
    for (int r = 0; r < 8; ++r) {
      int xr = r + half * 8;               // pixel row within tile
      float val = c[r];
      if (MODE == 1) val = gelu_exact(val);
      long addr = (obase + xr) * EMB + n;
      if (MODE == 2) {
        float dv = (float)D[addr];
        D[addr] = (_Float16)(dv + val);
      } else {
        Y[addr] = (_Float16)val;
      }
    }
  }
}

// ------------------------------------------------------- attention scores ---
// one wave per (win, head): 16x16 scores via 128 WMMA k-steps (K = 4096),
// then softmax through LDS, store probs as f16.
__global__ __launch_bounds__(32)
void k_scores(const _Float16* __restrict__ Q, const _Float16* __restrict__ K,
              _Float16* __restrict__ A) {
  __shared__ float Sm[16][16];
  int wid = blockIdx.x;                     // win*4 + head
  int win = wid >> 2, h = wid & 3;
  int lane = threadIdx.x;
  int half = lane >> 4, t = lane & 15;      // token index for A-row / B-col
  const _Float16* qb = Q + (long)(win * NTOK + t) * 256 * EMB + h * 16;
  const _Float16* kb = K + (long)(win * NTOK + t) * 256 * EMB + h * 16;
  v8f c;
  for (int r = 0; r < 8; ++r) c[r] = 0.0f;
#pragma unroll 4
  for (int kk = 0; kk < 128; ++kk) {
    long s0 = (long)kk * 2 * EMB;
    V16U a, b;
    a.h[0] = *(const v8h*)(qb + s0 + half * 8);          // spatial s0, ch half*8
    a.h[1] = *(const v8h*)(qb + s0 + EMB + half * 8);    // spatial s0+1
    b.h[0] = *(const v8h*)(kb + s0 + half * 8);          // spatial s0, ch half*8
    b.h[1] = *(const v8h*)(kb + s0 + EMB + half * 8);    // spatial s0+1
    c = __builtin_amdgcn_wmma_f32_16x16x32_f16(false, a.v, false, b.v,
                                               (short)0, c, false, false);
  }
  const float scale = 1.0f / (256.0f * 4.0f);   // 1/((sh*sw)*sqrt(dh))
  for (int r = 0; r < 8; ++r) Sm[r + half * 8][t] = c[r] * scale;
  __syncthreads();
  if (lane < 16) {
    float mx = -1e30f;
    for (int j = 0; j < 16; ++j) mx = fmaxf(mx, Sm[lane][j]);
    float e[16], s = 0.f;
    for (int j = 0; j < 16; ++j) { e[j] = __expf(Sm[lane][j] - mx); s += e[j]; }
    float inv = 1.0f / s;
    _Float16* arow = A + (long)(wid * 16 + lane) * 16;
    for (int j = 0; j < 16; ++j) arow[j] = (_Float16)(e[j] * inv);
  }
}

// ------------------------------------------------------------- o = A @ V ----
// thread = (win, head, out-token i, spatial s), produces 16 head-channels
__global__ __launch_bounds__(256)
void k_attnmul(const _Float16* __restrict__ A, const _Float16* __restrict__ V,
               _Float16* __restrict__ O) {
  long gid = (long)blockIdx.x * 256 + threadIdx.x;   // < NWIN*NHD*16*256
  int s = (int)(gid & 255);
  int i = (int)((gid >> 8) & 15);
  int wh = (int)(gid >> 12);
  int win = wh >> 2, h = wh & 3;
  const _Float16* ar = A + (long)(wh * 16 + i) * 16;
  float aw[16];
  for (int j = 0; j < 16; ++j) aw[j] = (float)ar[j];
  float acc[16];
  for (int c = 0; c < 16; ++c) acc[c] = 0.f;
  for (int j = 0; j < 16; ++j) {
    const _Float16* vb = V + ((long)(win * NTOK + j) * 256 + s) * EMB + h * 16;
    v8h v0 = *(const v8h*)(vb);
    v8h v1 = *(const v8h*)(vb + 8);
    float w = aw[j];
    for (int c = 0; c < 8; ++c) acc[c]     += w * (float)v0[c];
    for (int c = 0; c < 8; ++c) acc[c + 8] += w * (float)v1[c];
  }
  _Float16* ob = O + ((long)(win * NTOK + i) * 256 + s) * EMB + h * 16;
  v8h o0, o1;
  for (int c = 0; c < 8; ++c) { o0[c] = (_Float16)acc[c]; o1[c] = (_Float16)acc[c + 8]; }
  *(v8h*)ob = o0;
  *(v8h*)(ob + 8) = o1;
}

// ----------------------------------------------------------- output head ----
// 64 -> 128 (gelu) -> 1, un-window relayout, f32 output
__global__ __launch_bounds__(256)
void k_out(const _Float16* __restrict__ Dq, const float* __restrict__ W1,
           const float* __restrict__ b1, const float* __restrict__ W2,
           const float* __restrict__ b2, float* __restrict__ out) {
  __shared__ float W1l[HID * EMB];    // 32 KB
  __shared__ float W2l[HID];
  __shared__ float b1l[HID];
  __shared__ float b2s;
  int tid = threadIdx.x;
  for (int i = tid; i < HID * EMB; i += 256) W1l[i] = W1[i];
  if (tid < HID) { W2l[tid] = W2[tid]; b1l[tid] = b1[tid]; }
  if (tid == 0) b2s = b2[0];
  __syncthreads();

  long gid = (long)blockIdx.x * 256 + tid;
  int X = (int)(gid & 255), Yp = (int)((gid >> 8) & 255), b = (int)(gid >> 16);
  int syi = Yp >> 4, y = Yp & 15, sxi = X >> 4, x = X & 15;
  int win = (b * NWX + (syi >> 2)) * NWX + (sxi >> 2);
  int tok = (syi & 3) * WSZ + (sxi & 3);
  const _Float16* db = Dq + ((long)(win * NTOK + tok) * 256 + y * 16 + x) * EMB;
  float xv[EMB];
  for (int cc = 0; cc < 8; ++cc) {
    v8h pk = *(const v8h*)(db + cc * 8);
    for (int e = 0; e < 8; ++e) xv[cc * 8 + e] = (float)pk[e];
  }
  float acc = b2s;
  for (int o = 0; o < HID; ++o) {
    float hv = b1l[o];
    const float* wr = W1l + o * EMB;
    for (int c = 0; c < EMB; ++c) hv += wr[c] * xv[c];
    acc += W2l[o] * gelu_exact(hv);
  }
  out[(long)b * 65536 + (long)Yp * 256 + X] = acc;
}

// ---------------------------------------------------------------- driver ----
extern "C" void kernel_launch(void* const* d_in, const int* in_sizes, int n_in,
                              void* d_out, int out_size, void* d_ws, size_t ws_size,
                              hipStream_t stream) {
  const float* v   = (const float*)d_in[0];
  const float* iW1 = (const float*)d_in[1];
  const float* ib1 = (const float*)d_in[2];
  const float* iW2 = (const float*)d_in[3];
  const float* ib2 = (const float*)d_in[4];
  const float* Wq  = (const float*)d_in[5];
  const float* bq  = (const float*)d_in[6];
  const float* Wk  = (const float*)d_in[7];
  const float* bk  = (const float*)d_in[8];
  const float* Wv  = (const float*)d_in[9];
  const float* bv  = (const float*)d_in[10];
  const float* Wo  = (const float*)d_in[11];
  const float* bo  = (const float*)d_in[12];
  const float* mW1 = (const float*)d_in[13];
  const float* mb1 = (const float*)d_in[14];
  const float* mW2 = (const float*)d_in[15];
  const float* mb2 = (const float*)d_in[16];
  const float* oW1 = (const float*)d_in[17];
  const float* ob1 = (const float*)d_in[18];
  const float* oW2 = (const float*)d_in[19];
  const float* ob2 = (const float*)d_in[20];
  const int* dsy   = (const int*)d_in[21];
  const int* dsx   = (const int*)d_in[22];
  float* outp = (float*)d_out;

  size_t plane = (size_t)NPIX * EMB * sizeof(_Float16);  // 64 MB
  if (ws_size < 5 * plane + (size_t)NWIN * NHD * 256 * sizeof(_Float16)) return;
  char* ws = (char*)d_ws;
  _Float16* Dq = (_Float16*)(ws);
  _Float16* Xn = (_Float16*)(ws + plane);
  _Float16* Qb = (_Float16*)(ws + 2 * plane);
  _Float16* Kb = (_Float16*)(ws + 3 * plane);
  _Float16* Vb = (_Float16*)(ws + 4 * plane);
  _Float16* Ab = (_Float16*)(ws + 5 * plane);

  k_embed<<<NPIX / 256, 256, 0, stream>>>(v, iW1, ib1, iW2, ib2, dsy, dsx, Dq);

  const int gemmGrid = NPIX / 16 / 8;    // 4096 blocks * 8 waves * 16-row tiles
  for (int l = 0; l < NLAYER; ++l) {
    int shift = (l & 1) ? SHIFT : 0;
    int fwd = shift ? (NSUB - shift) : 0;   // roll(-shift)
    int inv = shift;                        // roll(+shift)
    const float* wq = Wq + (size_t)l * EMB * EMB;
    const float* wk = Wk + (size_t)l * EMB * EMB;
    const float* wv = Wv + (size_t)l * EMB * EMB;
    const float* wo = Wo + (size_t)l * EMB * EMB;
    const float* m1 = mW1 + (size_t)l * EMB * EMB;
    const float* m2 = mW2 + (size_t)l * EMB * EMB;

    // attention branch
    k_inorm<<<NWIN * NTOK, 64, 0, stream>>>(Dq, Xn, fwd);
    k_gemm<0><<<gemmGrid, 256, 0, stream>>>(Xn, wq, bq + l * EMB, Qb, nullptr, 0);
    k_gemm<0><<<gemmGrid, 256, 0, stream>>>(Xn, wk, bk + l * EMB, Kb, nullptr, 0);
    k_gemm<0><<<gemmGrid, 256, 0, stream>>>(Xn, wv, bv + l * EMB, Vb, nullptr, 0);
    k_scores<<<NWIN * NHD, 32, 0, stream>>>(Qb, Kb, Ab);
    k_attnmul<<<(NWIN * NHD * 16 * 256) / 256, 256, 0, stream>>>(Ab, Vb, Xn);
    k_gemm<2><<<gemmGrid, 256, 0, stream>>>(Xn, wo, bo + l * EMB, nullptr, Dq, inv);

    // MLP branch
    k_inorm<<<NWIN * NTOK, 64, 0, stream>>>(Dq, Xn, 0);
    k_gemm<1><<<gemmGrid, 256, 0, stream>>>(Xn, m1, mb1 + l * EMB, Qb, nullptr, 0);
    k_gemm<2><<<gemmGrid, 256, 0, stream>>>(Qb, m2, mb2 + l * EMB, nullptr, Dq, 0);
  }

  k_out<<<NPIX / 256, 256, 0, stream>>>(Dq, oW1, ob1, oW2, ob2, outp);
}
